// LocalizedFiltering_24077586662075
// MI455X (gfx1250) — compile-verified, matrix operand order checked by reading, not run on hardware
//
#include <hip/hip_runtime.h>
#include <hip/hip_bf16.h>

// ---------------- fixed problem sizes (match reference setup) ----------------
#define NTOK 65536      // N = BS * L
#define EDIM 1024       // E
#define HDIM 512        // E/2
#define SEQL 8192       // L (power of two -> (tok & 8191)==0 marks seq start)
#define NSEQ 8          // BS
#define EPSV 1e-6f

typedef __bf16 bf16;
typedef __attribute__((ext_vector_type(16))) __bf16 v16bf;
typedef __attribute__((ext_vector_type(8)))  __bf16 v8bf;
typedef __attribute__((ext_vector_type(4)))  __bf16 v4bf;
typedef __attribute__((ext_vector_type(8)))  float  v8f;

// Assemble a 16-element bf16 WMMA fragment from a padded LDS row.
// Lane l (l<16): row l%16, K in {0..7} U {16..23};  lane l>=16: K in {8..15} U {24..31}.
__device__ __forceinline__ v16bf load_frag(const bf16* rowbase, int hi) {
    v8bf lo = *(const v8bf*)(rowbase + hi * 8);
    v8bf hh = *(const v8bf*)(rowbase + 16 + hi * 8);
    return __builtin_shufflevector(lo, hh, 0,1,2,3,4,5,6,7,8,9,10,11,12,13,14,15);
}

// ---------------- prep: B^T bf16 layouts --------------------------------------
// Bt1 [512][2048]: Bt1[n][k] = W1[(k%1024)][ k<1024 ? n : n+512 ]   (W1: [1024,1024])
__global__ void k_prep_bt1(const float* __restrict__ w1, bf16* __restrict__ bt1) {
    int idx = blockIdx.x * 256 + threadIdx.x;          // 512*2048 total
    int n = idx >> 11, k = idx & 2047;
    float v = w1[(size_t)(k & 1023) * 1024 + ((k < 1024) ? n : n + 512)];
    bt1[idx] = (bf16)v;
}
// Bt2 [1024][1024]: Bt2[n][k] = W2[(k%512)][ k<512 ? n : n+1024 ]   (W2: [512,2048])
__global__ void k_prep_bt2(const float* __restrict__ w2, bf16* __restrict__ bt2) {
    int idx = blockIdx.x * 256 + threadIdx.x;          // 1024*1024 total
    int n = idx >> 10, k = idx & 1023;
    float v = w2[(size_t)(k & 511) * 2048 + ((k < 512) ? n : n + 1024)];
    bt2[idx] = (bf16)v;
}
// cache2 bf16 [8][512] = lf2_caches[pre_idx[s]]
__global__ void k_prep_cache2(const float* __restrict__ lf2, const int* __restrict__ preidx,
                              bf16* __restrict__ c2) {
    int idx = blockIdx.x * 256 + threadIdx.x;          // 4096 total
    int s = idx >> 9, k = idx & 511;
    c2[idx] = (bf16)lf2[(size_t)preidx[s] * 512 + k];
}

// ---------------- GEMM 1: out1[N,512] (bf16) ----------------------------------
// Effective A [N, 2048]: k<1024 -> prev token row (or lf1 cache at seq start),
//                        k>=1024 -> current token row.  A fp32 -> cvt bf16 -> LDS.
// Block tile 128(M) x 128(N), 8 waves as 4(M-groups of 32) x 2(N-groups of 64),
// per wave 2 A-frags x 4 B-frags = 8 WMMAs / K-step.  Double-buffered LDS.
__global__ __launch_bounds__(256) void k_gemm1(
    const float* __restrict__ X,     // inputs [N,1024] fp32
    const float* __restrict__ lf1,   // [16384,1024] fp32
    const int*   __restrict__ preidx,
    const bf16*  __restrict__ Bt1,   // [512,2048]
    const float* __restrict__ b1,    // [512]
    bf16* __restrict__ out1)         // [N,512]
{
    const int tidx = threadIdx.x;
    const int lane = tidx & 31, wv = tidx >> 5;
    const int mg = wv & 3, ng = wv >> 2;     // wave's 32-row / 64-col group
    const int tileN = blockIdx.x;            // 0..3
    const int r0 = blockIdx.y * 128;         // 512 row tiles

    __shared__ __align__(16) bf16 lA[2][128][40];
    __shared__ __align__(16) bf16 lB[2][128][40];

    v8f acc[2][4] = {};

    // --- loop-invariant A source pointers (4 row-slots per thread) ---
    const int arow = tidx >> 3;              // 0..31 (+32*i)
    const int akc  = (tidx & 7) * 4;         // float4 within 32-K chunk
    const float* aprev[4];
    const float* acur[4];
#pragma unroll
    for (int i = 0; i < 4; ++i) {
        const int tok = r0 + arow + i * 32;
        aprev[i] = ((tok & (SEQL - 1)) == 0)
                 ? lf1 + (size_t)preidx[tok >> 13] * EDIM + akc
                 : X + (size_t)(tok - 1) * EDIM + akc;
        acur[i]  = X + (size_t)tok * EDIM + akc;
    }
    // --- B source pointers (2 col-slots per thread) ---
    const bf16* bsrc[2];
    int bcol[2], bkcv[2];
#pragma unroll
    for (int s = 0; s < 2; ++s) {
        const int id = tidx + s * 256;       // 0..511
        bcol[s] = id >> 2;                   // 0..127
        bkcv[s] = (id & 3) * 8;
        bsrc[s] = Bt1 + (size_t)(tileN * 128 + bcol[s]) * 2048 + bkcv[s];
    }

    const int hr = lane & 15, hi = lane >> 4;
    const bf16* aRow0 = &lA[0][mg * 32 + hr][0];
    const bf16* aRow1 = &lA[0][mg * 32 + 16 + hr][0];
    const bf16* bRow  = &lB[0][ng * 64 + hr][0];
    const int bufStrideA = 128 * 40;         // elements between buffers

    float4 av[4];
    v8bf   bv[2];

    // tile loader: kb in [0,64); first 32 = prev half, last 32 = cur half
    auto loadTile = [&](int kb) {
        const bool pf = (kb < 32);
        const int  ko = (pf ? kb : kb - 32) * 32;
#pragma unroll
        for (int i = 0; i < 4; ++i)
            av[i] = *(const float4*)((pf ? aprev[i] : acur[i]) + ko);
#pragma unroll
        for (int s = 0; s < 2; ++s)
            bv[s] = *(const v8bf*)(bsrc[s] + kb * 32);
    };
    auto storeTile = [&](int nb) {
#pragma unroll
        for (int i = 0; i < 4; ++i) {
            v4bf s4;
            s4[0] = (bf16)av[i].x; s4[1] = (bf16)av[i].y;
            s4[2] = (bf16)av[i].z; s4[3] = (bf16)av[i].w;
            *(v4bf*)&lA[nb][arow + i * 32][akc] = s4;
        }
#pragma unroll
        for (int s = 0; s < 2; ++s)
            *(v8bf*)&lB[nb][bcol[s]][bkcv[s]] = bv[s];
    };

    loadTile(0);
    storeTile(0);
    for (int kb = 0; kb < 64; ++kb) {
        const int cur = kb & 1;
        __syncthreads();
        if (kb + 1 < 64) loadTile(kb + 1);
        // Load ALL fragments first so the 12 ds_load_b128 issue back-to-back
        // and WMMAs run under partial dscnt waits.
        const v16bf a0 = load_frag(aRow0 + cur * bufStrideA, hi);
        const v16bf a1 = load_frag(aRow1 + cur * bufStrideA, hi);
        v16bf bfr[4];
#pragma unroll
        for (int j = 0; j < 4; ++j)
            bfr[j] = load_frag(bRow + cur * bufStrideA + j * 16 * 40, hi);
#pragma unroll
        for (int j = 0; j < 4; ++j) {
            acc[0][j] = __builtin_amdgcn_wmma_f32_16x16x32_bf16(
                false, a0, false, bfr[j], (short)0, acc[0][j], false, false);
            acc[1][j] = __builtin_amdgcn_wmma_f32_16x16x32_bf16(
                false, a1, false, bfr[j], (short)0, acc[1][j], false, false);
        }
        if (kb + 1 < 64) storeTile(cur ^ 1);
    }

#pragma unroll
    for (int j = 0; j < 4; ++j) {
        const int col = tileN * 128 + ng * 64 + j * 16 + hr;
        const float bb = b1[col];
#pragma unroll
        for (int t = 0; t < 2; ++t)
#pragma unroll
            for (int v = 0; v < 8; ++v) {
                const int row = r0 + mg * 32 + t * 16 + hi * 8 + v;
                out1[(size_t)row * HDIM + col] = (bf16)(acc[t][j][v] + bb);
            }
    }
}

// ---------------- GEMM 2: d_out[N,1024] = out2 + b2 + inputs (pre-norm) -------
// Effective A [N, 1024] bf16: k<512 -> prev out1 row (or cache2 at seq start),
//                             k>=512 -> current out1 row.
__global__ __launch_bounds__(256) void k_gemm2(
    const bf16*  __restrict__ out1,  // [N,512]
    const bf16*  __restrict__ c2,    // [8,512]
    const bf16*  __restrict__ Bt2,   // [1024,1024]
    const float* __restrict__ b2,    // [1024]
    const float* __restrict__ X,     // inputs [N,1024] (residual)
    float* __restrict__ out)         // d_out [N,1024]
{
    const int tidx = threadIdx.x;
    const int lane = tidx & 31, wv = tidx >> 5;
    const int mg = wv & 3, ng = wv >> 2;
    const int tileN = blockIdx.x;            // 0..7
    const int r0 = blockIdx.y * 128;

    __shared__ __align__(16) bf16 lA[2][128][40];
    __shared__ __align__(16) bf16 lB[2][128][40];

    v8f acc[2][4] = {};

    const int arow = tidx >> 2;              // 0..63 (+64*i)
    const int akc  = (tidx & 3) * 8;         // 8 bf16
    const bf16* aprev[2];
    const bf16* acur[2];
#pragma unroll
    for (int i = 0; i < 2; ++i) {
        const int tok = r0 + arow + i * 64;
        aprev[i] = ((tok & (SEQL - 1)) == 0)
                 ? c2 + (size_t)(tok >> 13) * HDIM + akc
                 : out1 + (size_t)(tok - 1) * HDIM + akc;
        acur[i]  = out1 + (size_t)tok * HDIM + akc;
    }
    const bf16* bsrc[2];
    int bcol[2], bkcv[2];
#pragma unroll
    for (int s = 0; s < 2; ++s) {
        const int id = tidx + s * 256;
        bcol[s] = id >> 2;
        bkcv[s] = (id & 3) * 8;
        bsrc[s] = Bt2 + (size_t)(tileN * 128 + bcol[s]) * 1024 + bkcv[s];
    }

    const int hr = lane & 15, hi = lane >> 4;
    const bf16* aRow0 = &lA[0][mg * 32 + hr][0];
    const bf16* aRow1 = &lA[0][mg * 32 + 16 + hr][0];
    const bf16* bRow  = &lB[0][ng * 64 + hr][0];
    const int bufStrideA = 128 * 40;

    v8bf av[2], bv[2];

    auto loadTile = [&](int kb) {            // kb in [0,32); halves at 16
        const bool pf = (kb < 16);
        const int  ko = (pf ? kb : kb - 16) * 32;
#pragma unroll
        for (int i = 0; i < 2; ++i)
            av[i] = *(const v8bf*)((pf ? aprev[i] : acur[i]) + ko);
#pragma unroll
        for (int s = 0; s < 2; ++s)
            bv[s] = *(const v8bf*)(bsrc[s] + kb * 32);
    };
    auto storeTile = [&](int nb) {
#pragma unroll
        for (int i = 0; i < 2; ++i)
            *(v8bf*)&lA[nb][arow + i * 64][akc] = av[i];
#pragma unroll
        for (int s = 0; s < 2; ++s)
            *(v8bf*)&lB[nb][bcol[s]][bkcv[s]] = bv[s];
    };

    loadTile(0);
    storeTile(0);
    for (int kb = 0; kb < 32; ++kb) {
        const int cur = kb & 1;
        __syncthreads();
        if (kb + 1 < 32) loadTile(kb + 1);
        const v16bf a0 = load_frag(aRow0 + cur * bufStrideA, hi);
        const v16bf a1 = load_frag(aRow1 + cur * bufStrideA, hi);
        v16bf bfr[4];
#pragma unroll
        for (int j = 0; j < 4; ++j)
            bfr[j] = load_frag(bRow + cur * bufStrideA + j * 16 * 40, hi);
#pragma unroll
        for (int j = 0; j < 4; ++j) {
            acc[0][j] = __builtin_amdgcn_wmma_f32_16x16x32_bf16(
                false, a0, false, bfr[j], (short)0, acc[0][j], false, false);
            acc[1][j] = __builtin_amdgcn_wmma_f32_16x16x32_bf16(
                false, a1, false, bfr[j], (short)0, acc[1][j], false, false);
        }
        if (kb + 1 < 32) storeTile(cur ^ 1);
    }

#pragma unroll
    for (int j = 0; j < 4; ++j) {
        const int col = tileN * 128 + ng * 64 + j * 16 + hr;
        const float bb = b2[col];
#pragma unroll
        for (int t = 0; t < 2; ++t)
#pragma unroll
            for (int v = 0; v < 8; ++v) {
                const int row = r0 + mg * 32 + t * 16 + hi * 8 + v;
                const size_t o = (size_t)row * EDIM + col;
                out[o] = acc[t][j][v] + bb + X[o];
            }
    }
}

// ---------------- RMSNorm (in place on d_out) ---------------------------------
__global__ __launch_bounds__(256) void k_rmsnorm(float* __restrict__ y,
                                                 const float* __restrict__ lnw) {
    const int row = blockIdx.x;
    const int tid = threadIdx.x;
    float4 v = *(float4*)(y + (size_t)row * EDIM + tid * 4);
    float ss = v.x * v.x + v.y * v.y + v.z * v.z + v.w * v.w;
#pragma unroll
    for (int m = 16; m > 0; m >>= 1) ss += __shfl_xor(ss, m, 32);
    __shared__ float ps[8];
    if ((tid & 31) == 0) ps[tid >> 5] = ss;
    __syncthreads();
    float tot = 0.f;
#pragma unroll
    for (int i = 0; i < 8; ++i) tot += ps[i];
    const float scale = rsqrtf(tot * (1.0f / (float)EDIM) + EPSV);
    const float4 w = *(const float4*)(lnw + tid * 4);
    v.x *= scale * w.x; v.y *= scale * w.y; v.z *= scale * w.z; v.w *= scale * w.w;
    *(float4*)(y + (size_t)row * EDIM + tid * 4) = v;
}

// ---------------- host launcher -----------------------------------------------
extern "C" void kernel_launch(void* const* d_in, const int* in_sizes, int n_in,
                              void* d_out, int out_size, void* d_ws, size_t ws_size,
                              hipStream_t stream) {
    const float* inputs = (const float*)d_in[0];
    const float* w1     = (const float*)d_in[1];
    const float* w2     = (const float*)d_in[2];
    const float* b1     = (const float*)d_in[3];
    const float* b2     = (const float*)d_in[4];
    const float* lnw    = (const float*)d_in[5];
    const float* lf1    = (const float*)d_in[6];
    const float* lf2    = (const float*)d_in[7];
    const int*   preidx = (const int*)  d_in[8];
    float* out = (float*)d_out;

    char* ws = (char*)d_ws;
    bf16* Bt1    = (bf16*)(ws);                              // 512*2048*2  = 2 MB
    bf16* Bt2    = (bf16*)(ws + (2u << 20));                 // 1024*1024*2 = 2 MB
    bf16* cache2 = (bf16*)(ws + (4u << 20));                 // 8*512*2     = 8 KB
    bf16* out1   = (bf16*)(ws + (4u << 20) + (64u << 10));   // N*512*2     = 64 MB

    k_prep_bt1   <<<(512 * 2048) / 256, 256, 0, stream>>>(w1, Bt1);
    k_prep_bt2   <<<(1024 * 1024) / 256, 256, 0, stream>>>(w2, Bt2);
    k_prep_cache2<<<(NSEQ * 512) / 256, 256, 0, stream>>>(lf2, preidx, cache2);

    k_gemm1<<<dim3(HDIM / 128, NTOK / 128), 256, 0, stream>>>(inputs, lf1, preidx, Bt1, b1, out1);
    k_gemm2<<<dim3(EDIM / 128, NTOK / 128), 256, 0, stream>>>(out1, cache2, Bt2, b2, inputs, out);
    k_rmsnorm<<<NTOK, 256, 0, stream>>>(out, lnw);
}
